// ModelConTucker_46016279609474
// MI455X (gfx1250) — matrix operand order repaired and checked by reading, not compiled, not run on hardware
//
#include <hip/hip_runtime.h>
#include <hip/hip_bf16.h>

// CDNA5 / gfx1250 Tucker contraction:
//   out[j] = sum_{a,b,c} U[a,b,c] * Q0[i0(j),a] * Q1[i0(j),b] * Q2[i2(j),c]
// Per 16-sample tile (one wave32):
//   V(16x32) = K0(16x1024) x Umat(1024x32), K0[j,ab]=q0[j,a]*q1[j,b],
//   64 x v_wmma_f32_16x16x32_f16 (2 c-tiles x 32 K-steps), then
//   out[j] = sum_c V[j,c]*q2[j,c] via shfl_xor over lanes.
// U is staged once per block into LDS (f16, pre-swizzled to the exact
// per-lane WMMA B-fragment layout). An opaque per-tile "salt" (always 0,
// produced by inline asm the compiler cannot fold) is mixed into the LDS
// address so LICM cannot hoist the 64 fragments out of the tile loop --
// that hoisting caused catastrophic scratch spills in the previous build.

typedef __attribute__((ext_vector_type(16))) _Float16 v16h;
typedef __attribute__((ext_vector_type(8)))  float    v8f;
typedef __attribute__((ext_vector_type(4)))  float    v4f;

__device__ __forceinline__ int clampi(int v, int hi) {
    v = v < 0 ? 0 : v;
    return v > hi ? hi : v;
}

__global__ __launch_bounds__(256)
void tucker_wmma_kernel(const float* __restrict__ U,
                        const float* __restrict__ Q0,
                        const float* __restrict__ Q1,
                        const float* __restrict__ Q2,
                        const int*   __restrict__ x,
                        float*       __restrict__ out,
                        int n, int r0m1, int r1m1, int r2m1, int tiles)
{
    // Fragment f = kk*2 + ct (kk = K-chunk = a index, ct = c tile 0/1).
    // Lane L of fragment f holds 16 halves:
    //   element e -> U[kk*1024 + (e + 16*(L>>4))*32 + ct*16 + (L&15)]
    __shared__ _Float16 Ulds[32768];   // 64 KB

    for (int i = threadIdx.x; i < 32768; i += blockDim.x) {
        int f   = i >> 9;
        int rem = i & 511;
        int ln  = rem >> 4;
        int e   = rem & 15;
        int kk  = f >> 1, ct = f & 1;
        int hh  = ln >> 4, nn = ln & 15;
        int src = kk * 1024 + (e + 16 * hh) * 32 + ct * 16 + nn;
        Ulds[i] = (_Float16)U[src];
    }
    __syncthreads();

    const int lane = threadIdx.x & 31;
    const int hi   = lane >> 4;      // 16-lane half
    const int row  = lane & 15;      // sample row within tile / N column
    const int wid    = blockIdx.x * (blockDim.x >> 5) + (threadIdx.x >> 5);
    const int nwaves = gridDim.x * (blockDim.x >> 5);

    for (int tile = wid; tile < tiles; tile += nwaves) {
        const int base = tile * 16;

        // Opaque zero derived from `tile`: defeats LICM of the B-fragment
        // LDS loads across the tile loop (prevents register spilling).
        unsigned salt;
        asm volatile("v_xor_b32 %0, %1, %1" : "=v"(salt) : "v"(tile));
        const _Float16* __restrict__ Ub = Ulds + salt + lane * 16;

        // ---- gather q0 / q1 rows (reference uses idx[:,0] for BOTH) ----
        int j = base + row;
        if (j >= n) j = n - 1;
        const int xi0 = x[j * 3 + 0];
        const float* q0p = Q0 + (size_t)clampi(xi0, r0m1) * 32;
        const float* q1p = Q1 + (size_t)clampi(xi0, r1m1) * 32;

        // A-fragment k order chosen here (16x32, same as B in LDS): lanes 0-15 own K {0..15},
        // lanes 16-31 own K {16..31} -> float4 loads at {0,4,8,12}+16*hi.
        v4f t0 = *(const v4f*)(q1p + 16 * hi + 0);
        v4f t1 = *(const v4f*)(q1p + 16 * hi + 4);
        v4f t2 = *(const v4f*)(q1p + 16 * hi + 8);
        v4f t3 = *(const v4f*)(q1p + 16 * hi + 12);
        v16h q1h;
#pragma unroll
        for (int e = 0; e < 4; ++e) {
            q1h[e]      = (_Float16)t0[e];
            q1h[4 + e]  = (_Float16)t1[e];
            q1h[8 + e]  = (_Float16)t2[e];
            q1h[12 + e] = (_Float16)t3[e];
        }

        // Full q0 row as f16 scalars (broadcast factor per K-chunk: a = kk).
        _Float16 q0h[32];
#pragma unroll
        for (int g = 0; g < 8; ++g) {
            v4f q = *(const v4f*)(q0p + 4 * g);
            q0h[4 * g + 0] = (_Float16)q[0];
            q0h[4 * g + 1] = (_Float16)q[1];
            q0h[4 * g + 2] = (_Float16)q[2];
            q0h[4 * g + 3] = (_Float16)q[3];
        }

        // ---- V = K0 x Umat : 32 K-steps x 2 c-tiles of WMMA ----
        v8f acc0 = {};   // c = 0..15
        v8f acc1 = {};   // c = 16..31
#pragma unroll
        for (int kk = 0; kk < 32; ++kk) {
            v16h a  = q1h * q0h[kk];                       // 8x v_pk_mul_f16
            v16h b0 = *(const v16h*)(Ub + kk * 1024);      // frag (kk, ct=0)
            v16h b1 = *(const v16h*)(Ub + kk * 1024 + 512);// frag (kk, ct=1)
            acc0 = __builtin_amdgcn_wmma_f32_16x16x32_f16(false, a, false, b0,
                                                          (short)0, acc0, false, false);
            acc1 = __builtin_amdgcn_wmma_f32_16x16x32_f16(false, a, false, b1,
                                                          (short)0, acc1, false, false);
        }

        // ---- out[j] = sum_c V[j,c]*q2[j,c].  D-frag f32 layout: VGPR k holds
        // row (k + 8*hi), column = lane&15 (acc0: c, acc1: c+16). For fixed k
        // the x/q2 loads are half-wave-uniform rows -> broadcast + coalesced.
        #pragma unroll
        for (int k = 0; k < 8; ++k) {
            int r  = k + 8 * hi;
            int jj = base + r;
            int jc = jj >= n ? n - 1 : jj;
            int i2 = clampi(x[jc * 3 + 2], r2m1);
            const float* q2p = Q2 + (size_t)i2 * 32;
            float p = acc0[k] * q2p[row] + acc1[k] * q2p[16 + row];
            p += __shfl_xor(p, 1);
            p += __shfl_xor(p, 2);
            p += __shfl_xor(p, 4);
            p += __shfl_xor(p, 8);   // stays within each 16-lane half
            if (row == 0 && jj < n) out[jj] = p;
        }
    }
}

extern "C" void kernel_launch(void* const* d_in, const int* in_sizes, int n_in,
                              void* d_out, int out_size, void* d_ws, size_t ws_size,
                              hipStream_t stream) {
    const float* U  = (const float*)d_in[0];
    const float* Q0 = (const float*)d_in[1];
    const float* Q1 = (const float*)d_in[2];
    const float* Q2 = (const float*)d_in[3];
    const int*   x  = (const int*)d_in[4];
    float* out = (float*)d_out;

    const int n  = in_sizes[4] / 3;                 // number of samples
    const int r0 = in_sizes[1] / 32;                // table rows
    const int r1 = in_sizes[2] / 32;
    const int r2 = in_sizes[3] / 32;
    const int tiles = (n + 15) / 16;

    int blocks = (tiles + 7) / 8;                   // 8 waves per block
    if (blocks > 256) blocks = 256;                 // grid-stride beyond this
    if (blocks < 1) blocks = 1;

    tucker_wmma_kernel<<<blocks, 256, 0, stream>>>(U, Q0, Q1, Q2, x, out,
                                                   n, r0 - 1, r1 - 1, r2 - 1, tiles);
}
